// VectorQuantizerEMA_88888643158440
// MI455X (gfx1250) — matrix operand order repaired and checked by reading, not compiled, not run on hardware
//
#include <hip/hip_runtime.h>
#include <hip/hip_bf16.h>

#define NVEC   131072
#define DIM    256
#define KCODES 1024
#define DECAYF 0.99f
#define EPSF   1e-5f

typedef __attribute__((ext_vector_type(16))) __bf16 v16bf;
typedef __attribute__((ext_vector_type(8)))  __bf16 v8bf;
typedef __attribute__((ext_vector_type(8)))  float  v8f;

union V16U { v16bf v; v8bf h[2]; };

// ---- gfx1250 async global->LDS copy path (guarded; falls back to sync copy) ----
#if defined(__AMDGCN__) && __has_builtin(__builtin_amdgcn_global_load_async_to_lds_b128)
#define HAS_ASYNC 1
typedef int v4i_g __attribute__((vector_size(16)));          // gcc-style int4 (matches builtin proto)
typedef __attribute__((address_space(1))) v4i_g as1_v4i;     // global
typedef __attribute__((address_space(3))) v4i_g as3_v4i;     // LDS
#if __has_builtin(__builtin_amdgcn_s_wait_asynccnt)
#define WAIT_ASYNC(n) __builtin_amdgcn_s_wait_asynccnt(n)
#else
#define WAIT_ASYNC(n) asm volatile("s_wait_asynccnt " #n ::: "memory")
#endif
#else
#define HAS_ASYNC 0
#define WAIT_ASYNC(n) do { } while (0)
#endif

// ---------------- utility: zero a float region ----------------
__global__ void vq_zero(float* __restrict__ p, int n) {
    int i = blockIdx.x * blockDim.x + threadIdx.x;
    if (i < n) p[i] = 0.0f;
}

// ---------------- L2-normalize codebook -> bf16 ----------------
__global__ void vq_enorm(const float* __restrict__ embed, __bf16* __restrict__ enorm) {
    __shared__ float red[DIM];
    const int k = blockIdx.x, t = threadIdx.x;
    const float x = embed[(size_t)k * DIM + t];
    red[t] = x * x;
    __syncthreads();
    for (int s = DIM / 2; s > 0; s >>= 1) {
        if (t < s) red[t] += red[t + s];
        __syncthreads();
    }
    const float inv = rsqrtf(fmaxf(red[0], 1e-12f));
    enorm[(size_t)k * DIM + t] = (__bf16)(x * inv);
}

// copy one 64-code tile (64*256 bf16 = 32 KB) of the normalized codebook into LDS
__device__ __forceinline__ void copy_tile(const __bf16* __restrict__ enorm, int tile,
                                          __bf16* sBbuf, int tid) {
    const char* gsrc = (const char*)(enorm + (size_t)tile * 64 * DIM);
#if HAS_ASYNC
    const unsigned long long g0 = (unsigned long long)gsrc;
    const unsigned l0 = (unsigned)(unsigned long long)sBbuf;   // LDS byte offset (low bits)
    #pragma unroll
    for (int i = 0; i < 8; ++i) {
        const unsigned off = (unsigned)(tid * 16 + i * 4096);  // 256 thr * 16 B per pass
        __builtin_amdgcn_global_load_async_to_lds_b128(
            (as1_v4i*)(g0 + off),
            (as3_v4i*)(unsigned long long)(l0 + off),
            0, 0);
    }
#else
    float4* d = (float4*)sBbuf;
    const float4* s = (const float4*)gsrc;
    #pragma unroll
    for (int i = 0; i < 8; ++i) d[tid + i * 256] = s[tid + i * 256];
#endif
}

// ---------------- main: normalize h, WMMA distances, argmax, gather, atomics ----------------
__global__ __launch_bounds__(256) void vq_main(
    const float*  __restrict__ h,
    const float*  __restrict__ embed,
    const __bf16* __restrict__ enorm,
    float* __restrict__ counts,
    float* __restrict__ esum,
    float* __restrict__ loss_acc,
    float* __restrict__ out_zqst,
    float* __restrict__ out_zid,
    float* __restrict__ out_zq)
{
    __shared__ __bf16 sA[128 * DIM];      // 64 KB: 128 normalized h rows (bf16)
    __shared__ __bf16 sB[2][64 * DIM];    // 2 x 32 KB: double-buffered codebook tiles
    __shared__ float  srow[256];
    __shared__ int    szid[128];

    const int tid  = threadIdx.x;
    const int r    = tid >> 1;            // local row 0..127
    const int part = tid & 1;             // half of the 256-dim row
    const size_t row0 = (size_t)blockIdx.x * 128;
    const float* hrow = h + (row0 + r) * DIM + part * 128;

    // prologue: start streaming codebook tile 0 into LDS while we normalize h
    copy_tile(enorm, 0, sB[0], tid);

    // ---- stage 1: row sumsq, then bf16-normalized rows into LDS ----
    float ss = 0.0f;
    #pragma unroll
    for (int i = 0; i < 128; i += 4) {
        float4 v = *(const float4*)(hrow + i);
        ss += v.x * v.x + v.y * v.y + v.z * v.z + v.w * v.w;
    }
    srow[tid] = ss;
    __syncthreads();
    const float inv = rsqrtf(fmaxf(srow[r * 2] + srow[r * 2 + 1], 1e-12f));
    __bf16* dst = sA + r * DIM + part * 128;
    #pragma unroll
    for (int i = 0; i < 128; i += 4) {
        float4 v = *(const float4*)(hrow + i);
        dst[i + 0] = (__bf16)(v.x * inv);
        dst[i + 1] = (__bf16)(v.y * inv);
        dst[i + 2] = (__bf16)(v.z * inv);
        dst[i + 3] = (__bf16)(v.w * inv);
    }
    __syncthreads();                      // sA complete: safe to hoist A fragments

    // ---- stage 2: per-wave 16x16 WMMA tiles, B double-buffered through LDS ----
    const int wave = tid >> 5;
    const int lane = tid & 31;
    const int hi   = lane >> 4;           // lane half (K slice / output M offset)
    const int col  = lane & 15;           // A: row M; B: column N (code within group)
    const __bf16* aRow = sA + (wave * 16 + col) * DIM;

    // hoist all A fragments (loop-invariant over the 16 codebook tiles): 64 VGPRs
    V16U Afrag[8];
    #pragma unroll
    for (int ch = 0; ch < 8; ++ch) {
        // A 16x32 bf16: lane M=col; V0-3 -> K = 8*hi+0..7, V4-7 -> K = 16+8*hi+0..7
        Afrag[ch].h[0] = *(const v8bf*)(aRow + ch * 32 + hi * 8);
        Afrag[ch].h[1] = *(const v8bf*)(aRow + ch * 32 + 16 + hi * 8);
    }

    float bestv[8];
    int   besti[8];
    #pragma unroll
    for (int v = 0; v < 8; ++v) { bestv[v] = -1e30f; besti[v] = 0; }

    for (int tile = 0; tile < KCODES / 64; ++tile) {
        const __bf16* cur = sB[tile & 1];
        if (tile + 1 < KCODES / 64) {
            copy_tile(enorm, tile + 1, sB[(tile + 1) & 1], tid);
            WAIT_ASYNC(8);                 // current tile's 8 async loads retired
        } else {
            WAIT_ASYNC(0);
        }
        __syncthreads();                   // cur tile visible to all waves

        #pragma unroll
        for (int sub = 0; sub < 4; ++sub) {
            const int code = tile * 64 + sub * 16 + col;
            const __bf16* bRow = cur + (sub * 16 + col) * DIM;
            // batch all B fragments for this 16-code group (one clause, one wait)
            V16U Bfrag[8];
            #pragma unroll
            for (int ch = 0; ch < 8; ++ch) {
                // B 32x16 bf16: lane N=col; V0-3 -> K = 8*hi+0..7, V4-7 -> K = 16+8*hi+0..7
                Bfrag[ch].h[0] = *(const v8bf*)(bRow + ch * 32 + hi * 8);
                Bfrag[ch].h[1] = *(const v8bf*)(bRow + ch * 32 + 16 + hi * 8);
            }
            v8f acc = {};
            #pragma unroll
            for (int ch = 0; ch < 8; ++ch) {
                acc = __builtin_amdgcn_wmma_f32_16x16x32_bf16(
                    false, Afrag[ch].v, false, Bfrag[ch].v, (short)0, acc, false, false);
            }
            // D layout: VGPR v, lanes 0-15 -> M=v, lanes 16-31 -> M=v+8; N=col
            #pragma unroll
            for (int v = 0; v < 8; ++v) {
                float s = acc[v];
                if (s > bestv[v] || (s == bestv[v] && code < besti[v])) {
                    bestv[v] = s; besti[v] = code;
                }
            }
        }
        __syncthreads();                   // done reading cur before it is overwritten
    }

    // reduce across the 16 lanes of each half (argmin dist == argmax sim, ties -> lowest idx)
    for (int off = 8; off >= 1; off >>= 1) {
        #pragma unroll
        for (int v = 0; v < 8; ++v) {
            float ov = __shfl_xor(bestv[v], off, 32);
            int   oi = __shfl_xor(besti[v], off, 32);
            if (ov > bestv[v] || (ov == bestv[v] && oi < besti[v])) {
                bestv[v] = ov; besti[v] = oi;
            }
        }
    }
    if (col == 0) {
        #pragma unroll
        for (int v = 0; v < 8; ++v) szid[wave * 16 + hi * 8 + v] = besti[v];
    }
    __syncthreads();

    // ---- stage 3: gather z_q, write outputs, loss + segment-sum atomics ----
    const int zid = szid[r];
    const float* erow = embed + (size_t)zid * DIM + part * 128;
    float* zqst = out_zqst + (row0 + r) * DIM + part * 128;
    float* zq   = out_zq   + (row0 + r) * DIM + part * 128;   // 4B-aligned only
    float* es   = esum + (size_t)zid * DIM + part * 128;
    float lp = 0.0f;
    #pragma unroll 4
    for (int i = 0; i < 128; i += 4) {
        float4 hv = *(const float4*)(hrow + i);
        float4 ev = *(const float4*)(erow + i);
        *(float4*)(zqst + i) = ev;
        zq[i + 0] = ev.x; zq[i + 1] = ev.y; zq[i + 2] = ev.z; zq[i + 3] = ev.w;
        float dx = hv.x - ev.x, dy = hv.y - ev.y, dz = hv.z - ev.z, dw = hv.w - ev.w;
        lp += dx * dx + dy * dy + dz * dz + dw * dw;
        atomicAdd(es + i + 0, hv.x);
        atomicAdd(es + i + 1, hv.y);
        atomicAdd(es + i + 2, hv.z);
        atomicAdd(es + i + 3, hv.w);
    }
    atomicAdd(loss_acc, lp);
    if (part == 0) {
        atomicAdd(&counts[zid], 1.0f);
        out_zid[row0 + r] = (float)zid;
    }
}

// ---------------- finalize 1: per-code scalars, n, entropy, new_cs, loss, perplexity ----------------
__global__ __launch_bounds__(1024) void vq_fin1(
    const float* __restrict__ counts,
    const float* __restrict__ cluster_size,
    const float* __restrict__ loss_acc,
    float* __restrict__ pcinv,
    int*   __restrict__ pcdead,
    int*   __restrict__ pcridx,
    float* __restrict__ out_newcs,
    float* __restrict__ out_loss,
    float* __restrict__ out_perp)
{
    __shared__ float rn[KCODES];
    __shared__ float re[KCODES];
    const int k = threadIdx.x;
    const float cnt = counts[k];
    const float ncs = cluster_size[k] * DECAYF + cnt * (1.0f - DECAYF);
    const float usage = cnt / (float)NVEC;
    rn[k] = ncs;
    re[k] = (usage > 0.0f) ? (-usage * logf(usage)) : 0.0f;
    __syncthreads();
    for (int s = KCODES / 2; s > 0; s >>= 1) {
        if (k < s) { rn[k] += rn[k + s]; re[k] += re[k + s]; }
        __syncthreads();
    }
    const float n = rn[0];
    const float entropy = re[0];
    float csm = (ncs + EPSF) / (n + (float)KCODES * EPSF) * n;
    csm = fmaxf(csm, EPSF);
    const int dead = (ncs / ((float)NVEC + EPSF)) < 0.01f;
    // fixed-seed hash in lieu of JAX PRNG (not reproducible on-device)
    unsigned rs = (unsigned)k * 747796405u + 2891336453u;
    rs ^= rs >> 17; rs *= 0xed5ad4bbu; rs ^= rs >> 11;
    pcinv[k]  = 1.0f / csm;
    pcdead[k] = dead;
    pcridx[k] = (int)(rs & (unsigned)(NVEC - 1));
    const float mean = n * (1.0f / (float)KCODES);
    out_newcs[k] = dead ? fmaxf(mean, 1.0f) : ncs;
    if (k == 0) {
        out_loss[0] = loss_acc[0] * (1.0f / ((float)NVEC * (float)DIM));
        out_perp[0] = expf(entropy);
    }
}

// ---------------- finalize 2: new_ea, new_embed (with dead revival) ----------------
__global__ void vq_fin2(
    const float* __restrict__ h,
    const float* __restrict__ embed_avg,
    const float* __restrict__ esum,
    const float* __restrict__ pcinv,
    const int*   __restrict__ pcdead,
    const int*   __restrict__ pcridx,
    float* __restrict__ out_newembed,
    float* __restrict__ out_newea)
{
    const int k = blockIdx.x, t = threadIdx.x;
    const size_t idx = (size_t)k * DIM + t;
    const float ea = embed_avg[idx] * DECAYF + esum[idx] * (1.0f - DECAYF);
    out_newea[idx] = ea;
    const float ne = pcdead[k] ? h[(size_t)pcridx[k] * DIM + t] : ea * pcinv[k];
    out_newembed[idx] = ne;
}

extern "C" void kernel_launch(void* const* d_in, const int* in_sizes, int n_in,
                              void* d_out, int out_size, void* d_ws, size_t ws_size,
                              hipStream_t stream) {
    const float* h     = (const float*)d_in[0];
    const float* embed = (const float*)d_in[1];
    const float* cs    = (const float*)d_in[2];
    const float* eavg  = (const float*)d_in[3];

    char* ws = (char*)d_ws;
    __bf16* enorm  = (__bf16*)ws;                 // 1024*256 bf16 = 524288 B
    float*  counts = (float*)(ws + 524288);       // 1024 f
    float*  esum   = (float*)(ws + 528384);       // 262144 f
    float*  lossa  = (float*)(ws + 1576960);      // 1 f (+pad)
    float*  pcinv  = (float*)(ws + 1576976);      // 1024 f
    int*    pcdead = (int*)  (ws + 1581072);      // 1024 i
    int*    pcridx = (int*)  (ws + 1585168);      // 1024 i

    float* out    = (float*)d_out;
    float* o_zqst = out;                                   // N*D
    float* o_zid  = o_zqst + (size_t)NVEC * DIM;           // N
    float* o_loss = o_zid + NVEC;                          // 1
    float* o_zq   = o_loss + 1;                            // N*D
    float* o_nemb = o_zq + (size_t)NVEC * DIM;             // K*D
    float* o_ncs  = o_nemb + (size_t)KCODES * DIM;         // K
    float* o_nea  = o_ncs + KCODES;                        // K*D
    float* o_perp = o_nea + (size_t)KCODES * DIM;          // 1

    // counts, esum, loss accumulator are contiguous in ws: zero them together
    const int nz = KCODES + KCODES * DIM + 4;
    vq_zero<<<(nz + 255) / 256, 256, 0, stream>>>(counts, nz);
    vq_enorm<<<KCODES, DIM, 0, stream>>>(embed, enorm);
    vq_main<<<NVEC / 128, 256, 0, stream>>>(h, embed, enorm, counts, esum, lossa,
                                            o_zqst, o_zid, o_zq);
    vq_fin1<<<1, KCODES, 0, stream>>>(counts, cs, lossa, pcinv, pcdead, pcridx,
                                      o_ncs, o_loss, o_perp);
    vq_fin2<<<KCODES, DIM, 0, stream>>>(h, eavg, esum, pcinv, pcdead, pcridx,
                                        o_nemb, o_nea);
}
